// custom_Conv_Net_919123001432
// MI455X (gfx1250) — matrix operand order repaired and checked by reading, not compiled, not run on hardware
//
#include <hip/hip_runtime.h>

// ---------------------------------------------------------------------------
// CDNA5 (gfx1250, wave32) implementation of:
//   x = relu(bn(conv3x3(img)));  f = relu(bn(conv3x3(face)));
//   res = per-sample VALID corr(x, f);  softmax over H then W.
// Convs are implicit-GEMM on v_wmma_f32_16x16x32_bf16, tap-major K ordering,
// LDS double-buffering (one barrier per K-step), 64x64 per-wave register
// tiles, and TDM (tensor_load_to_lds) staging of the weight tile when the
// toolchain exposes the builtin.
// ---------------------------------------------------------------------------

typedef __attribute__((ext_vector_type(16))) __bf16 v16bf;
typedef __attribute__((ext_vector_type(8)))  float  v8f;

#define C_CH   512
#define KTOT   4608          // 9 taps * 512 channels
#define KSTEP  32
#define BM     128
#define BN     256
#define KITERS (KTOT / KSTEP)   // 144 = 9 taps * 16 channel-blocks

#if __has_builtin(__builtin_amdgcn_tensor_load_to_lds) && \
    __has_builtin(__builtin_amdgcn_s_wait_tensorcnt)
#define USE_TDM 1
#endif

#ifdef USE_TDM
typedef __attribute__((ext_vector_type(4))) unsigned int tdm_v4u;
typedef __attribute__((ext_vector_type(8))) int          tdm_v8i;
typedef __attribute__((ext_vector_type(4))) int          tdm_v4i;

// Issue one TDM 2D tile load: nRows x 32 bf16, row stride KTOT elements,
// packed contiguously into LDS at ldsDst. Wave-level op (EXEC ignored);
// caller gates to a single wave. Completion via s_wait_tensorcnt.
__device__ __forceinline__ void tdm_load_b_tile(const unsigned short* gsrc,
                                                void* ldsDst, int nRows) {
  unsigned long long ga = (unsigned long long)(size_t)gsrc;  // global byte addr
  unsigned int lds = (unsigned int)(size_t)ldsDst;           // LDS byte addr
  tdm_v4u g0;
  g0[0] = 1u;                                        // count=1, user descriptor
  g0[1] = lds;                                       // lds_addr [63:32]
  g0[2] = (unsigned int)(ga & 0xFFFFFFFFu);          // global_addr [95:64]
  g0[3] = (unsigned int)((ga >> 32) & 0x01FFFFFFu)   // global_addr [120:96]
        | (2u << 30);                                // type=2 ("image") [127:126]
  tdm_v8i g1;
  g1[0] = (int)(1u << 16);                           // data_size=1 (2 bytes)
  g1[1] = (int)((unsigned)KTOT << 16);               // tensor_dim0 lo16 @ [63:48]
  g1[2] = (int)(((unsigned)KTOT >> 16) |             // tensor_dim0 hi16
                ((unsigned)C_CH << 16));             // tensor_dim1 lo16
  g1[3] = (int)((unsigned)KSTEP << 16);              // tile_dim0=32 @ [127:112]
  g1[4] = (int)((unsigned)nRows & 0xFFFFu);          // tile_dim1; tile_dim2=0
  g1[5] = (int)KTOT;                                 // tensor_dim0_stride lo32
  g1[6] = 0;                                         // stride hi16; dim1_stride=0
  g1[7] = 0;
  tdm_v4i g2 = (tdm_v4i)0;                           // 2D tile: dims 2/3 unused
  tdm_v4i g3 = (tdm_v4i)0;
#if __clang_major__ >= 23
  tdm_v8i g4 = (tdm_v8i)0;
  __builtin_amdgcn_tensor_load_to_lds(g0, g1, g2, g3, g4, 0);
#else
  __builtin_amdgcn_tensor_load_to_lds(g0, g1, g2, g3, 0);
#endif
}
#endif  // USE_TDM

__device__ __forceinline__ unsigned short f32_to_bf16(float f) {
  unsigned int u = __float_as_uint(f);
  unsigned int r = u + 0x7FFFu + ((u >> 16) & 1u);   // round-to-nearest-even
  return (unsigned short)(r >> 16);
}

__global__ void cvt_f32_bf16(const float* __restrict__ in,
                             unsigned short* __restrict__ out, int n) {
  int i = blockIdx.x * blockDim.x + threadIdx.x;
  if (i < n) out[i] = f32_to_bf16(in[i]);
}

// Weights [O][I][3][3] -> bf16 permuted to tap-major [O][tap(9)][I]
__global__ void cvt_weight_tapmajor(const float* __restrict__ w,
                                    unsigned short* __restrict__ out, int n) {
  int idx = blockIdx.x * blockDim.x + threadIdx.x;    // over O*I*9
  if (idx < n) {
    int o   = idx / (C_CH * 9);
    int rem = idx - o * (C_CH * 9);
    int i   = rem / 9;
    int t   = rem - i * 9;
    out[((size_t)o * 9 + t) * C_CH + i] = f32_to_bf16(w[idx]);
  }
}

// Fold conv-bias + eval-BN into per-channel affine: y = conv*A[c] + B[c]
__global__ void prep_bn(const float* __restrict__ bias, const float* __restrict__ g,
                        const float* __restrict__ be,   const float* __restrict__ mean,
                        const float* __restrict__ var,
                        float* __restrict__ A, float* __restrict__ B) {
  int c = threadIdx.x;                       // 512 threads
  float s = g[c] * rsqrtf(var[c] + 1e-5f);
  A[c] = s;
  B[c] = (bias[c] - mean[c]) * s + be[c];
}

// ---------------------------------------------------------------------------
// Implicit-GEMM conv3x3 SAME + fused affine(BN) + ReLU.
//   A (im2col of input, bf16, tap-major K) : M = Bn*H*W, K = 4608
//   B^T (weights bf16, [O][tap][I])        : [N=512][K=4608]
//   out (f32 NCHW)
// Block: 256 thr = 8 waves, tile 128x256; wave tile 64x64 = 16 accumulators.
// Double-buffered LDS staging, one workgroup barrier per K-step. B tile via
// TDM when available.
// ---------------------------------------------------------------------------
template<int H, int W>
__global__ __launch_bounds__(256)
void conv3x3_bn_relu_wmma(const unsigned short* __restrict__ Abf,
                          const unsigned short* __restrict__ Wbf,
                          const float* __restrict__ bnA,
                          const float* __restrict__ bnB,
                          float* __restrict__ out,
                          int Bn) {
  constexpr int HW = H * W;
  const int Mtot = Bn * HW;

  __shared__ __align__(16) unsigned short As[2][BM * KSTEP];   // 2 x  8 KB
  __shared__ __align__(16) unsigned short Bs[2][BN * KSTEP];   // 2 x 16 KB

  const int tid   = threadIdx.x;
  const int mBase = blockIdx.y * BM;
  const int nBase = blockIdx.x * BN;

  const int waveId = tid >> 5;
  const int lane   = tid & 31;
  const int half   = lane >> 4;      // lane half selects K sub-range (ISA layout)
  const int ml     = lane & 15;
  const int waveM  = waveId & 1;     // 2 wave-rows  (64 M each)
  const int waveN  = waveId >> 1;    // 4 wave-cols  (64 N each)

  // --- per-thread A-tile loader coordinates: 16 channels of one row ---
  const int rowA   = tid >> 1;              // 0..127
  const int kHalfA = (tid & 1) * 16;        // which 16 channels of the K-step
  const int mRow   = mBase + rowA;
  const bool mOK   = (mRow < Mtot);
  int ab = 0, aoh = 0, aow = 0;
  if (mOK) {
    ab = mRow / HW;                          // compile-time-constant divisors
    int rem = mRow - ab * HW;
    aoh = rem / W;
    aow = rem - aoh * W;
  }
  const unsigned short* aImg = Abf + (size_t)ab * C_CH * HW;   // image base

  // Stage the A side of K-step kIter into LDS buffer `buf`.
  auto stageA = [&](int buf, int kIter) {
    const int tap = kIter >> 4;              // 0..8
    const int cb  = (kIter & 15) << 5;       // channel base, 0..480
    const int r   = tap / 3;
    const int s   = tap - r * 3;
    const int ih  = aoh + r - 1;
    const int iw  = aow + s - 1;
    const bool valid = mOK && ((unsigned)ih < (unsigned)H)
                           && ((unsigned)iw < (unsigned)W);
    unsigned int pk[8];
    if (valid) {
      // 16 loads, stride HW elements, immediate offsets off one base register
      const unsigned short* ap = aImg + (size_t)(cb + kHalfA) * HW + ih * W + iw;
      #pragma unroll
      for (int p = 0; p < 8; p++) {
        unsigned int lo = ap[(2 * p)     * HW];
        unsigned int hi = ap[(2 * p + 1) * HW];
        pk[p] = lo | (hi << 16);
      }
      __builtin_prefetch((const char*)ap + (size_t)KSTEP * HW * 2, 0, 1);
    } else {
      #pragma unroll
      for (int p = 0; p < 8; p++) pk[p] = 0u;   // SAME-padding zero fill
    }
    uint4* ad = (uint4*)(&As[buf][rowA * KSTEP + kHalfA]);
    ad[0] = make_uint4(pk[0], pk[1], pk[2], pk[3]);
    ad[1] = make_uint4(pk[4], pk[5], pk[6], pk[7]);
  };

  // Stage the B side of K-step kIter into LDS buffer `buf`.
  auto stageB = [&](int buf, int kIter) {
    const int tap = kIter >> 4;
    const int cb  = (kIter & 15) << 5;
#ifdef USE_TDM
    if (waveId == 0)
      tdm_load_b_tile(Wbf + (size_t)nBase * KTOT + tap * C_CH + cb,
                      &Bs[buf][0], BN);
#else
    const int nRow = tid;                    // 0..255: one 64-B row each
    const uint4* bsrc = (const uint4*)(Wbf + (size_t)(nBase + nRow) * KTOT
                                       + tap * C_CH + cb);
    uint4* bd = (uint4*)(&Bs[buf][nRow * KSTEP]);
    bd[0] = bsrc[0];
    bd[1] = bsrc[1];
    bd[2] = bsrc[2];
    bd[3] = bsrc[3];
    __builtin_prefetch((const char*)bsrc + 64, 0, 1);
#endif
  };

  v8f acc[4][4];
  #pragma unroll
  for (int i = 0; i < 4; i++)
    #pragma unroll
    for (int j = 0; j < 4; j++)
      acc[i][j] = (v8f){0.f,0.f,0.f,0.f,0.f,0.f,0.f,0.f};

  stageA(0, 0);
  stageB(0, 0);
  int buf = 0;
  for (int kIter = 0; kIter < KITERS; ++kIter) {
#ifdef USE_TDM
    if (waveId == 0) __builtin_amdgcn_s_wait_tensorcnt(0);  // TDM done pre-barrier
#endif
    __syncthreads();                         // staged tile `buf` published
    if (kIter + 1 < KITERS) {
      stageA(buf ^ 1, kIter + 1);
      stageB(buf ^ 1, kIter + 1);
    }

    union Frag { uint4 u[2]; v16bf v; };
    Frag aF[4], bF[4];
    #pragma unroll
    for (int i = 0; i < 4; i++) {
      int row = waveM * 64 + i * 16 + ml;
      const uint4* p = (const uint4*)(&As[buf][row * KSTEP]);
      aF[i].u[0] = p[half];                  // K {0-7} or {8-15}
      aF[i].u[1] = p[2 + half];              // K {16-23} or {24-31}
    }
    #pragma unroll
    for (int j = 0; j < 4; j++) {
      int col = waveN * 64 + j * 16 + ml;
      const uint4* p = (const uint4*)(&Bs[buf][col * KSTEP]);
      bF[j].u[0] = p[half];                  // K {0-7} or {8-15}
      bF[j].u[1] = p[2 + half];              // K {16-23} or {24-31}
    }
    #pragma unroll
    for (int i = 0; i < 4; i++)
      #pragma unroll
      for (int j = 0; j < 4; j++)
        acc[i][j] = __builtin_amdgcn_wmma_f32_16x16x32_bf16(
            false, aF[i].v, false, bF[j].v, (short)0, acc[i][j], false, false);
    buf ^= 1;
  }

  // ---- epilogue: affine (bias+BN folded) + ReLU, scatter to NCHW f32 ----
  #pragma unroll
  for (int j = 0; j < 4; j++) {
    int n = nBase + waveN * 64 + j * 16 + ml;     // output channel
    float sA = bnA[n];
    float sB = bnB[n];
    #pragma unroll
    for (int i = 0; i < 4; i++) {
      #pragma unroll
      for (int r = 0; r < 8; r++) {
        int m = mBase + waveM * 64 + i * 16 + half * 8 + r;  // C-layout VGPR r -> M
        if (m < Mtot) {
          int b   = m / HW;
          int rem = m - b * HW;
          int oh  = rem / W;
          int ow  = rem - oh * W;
          float val = acc[i][j][r] * sA + sB;
          val = val > 0.f ? val : 0.f;
          out[(((size_t)(b * C_CH + n)) * H + oh) * W + ow] = val;
        }
      }
    }
  }
}

// ---------------------------------------------------------------------------
// Per-sample VALID cross-correlation: res[b,ho,wo] = sum_{c,kh,kw} x*f.
// One block per (b, ho); 15 wo-accumulators/thread; wave-shuffle + LDS reduce.
// 2.9 GFLOP total; all re-reads hit the 192 MB L2.
// ---------------------------------------------------------------------------
__global__ __launch_bounds__(256)
void corr_kernel(const float* __restrict__ x,   // [64,512,28,28]
                 const float* __restrict__ f,   // [64,512,14,14]
                 float* __restrict__ res) {     // [64,15,15]
  const int b  = blockIdx.y;
  const int ho = blockIdx.x;
  const int tid = threadIdx.x;

  float acc[15];
  #pragma unroll
  for (int w = 0; w < 15; w++) acc[w] = 0.f;

  const int KT = 512 * 196;
  const float* xb = x + (size_t)b * 512 * 784;
  const float* fb = f + (size_t)b * 512 * 196;

  for (int k = tid; k < KT; k += 256) {
    int c   = k / 196;
    int rem = k - c * 196;
    int kh  = rem / 14;
    int kw  = rem - kh * 14;
    float fv = fb[k];
    const float* xr = xb + ((size_t)c * 28 + (ho + kh)) * 28 + kw;
    #pragma unroll
    for (int w = 0; w < 15; w++) acc[w] += fv * xr[w];
  }

  __shared__ float wsum[8][15];
  const int lane = tid & 31, wv = tid >> 5;
  #pragma unroll
  for (int w = 0; w < 15; w++) {
    float v = acc[w];
    #pragma unroll
    for (int off = 16; off > 0; off >>= 1)
      v += __shfl_down(v, (unsigned)off, 32);
    if (lane == 0) wsum[wv][w] = v;
  }
  __syncthreads();
  if (tid < 15) {
    float s = 0.f;
    #pragma unroll
    for (int q = 0; q < 8; q++) s += wsum[q][tid];
    res[(b * 15 + ho) * 15 + tid] = s;
  }
}

// softmax over H (axis 2), then softmax over W (axis 3) of that result.
__global__ void softmax_hw(const float* __restrict__ res, float* __restrict__ out) {
  const int b = blockIdx.x;
  __shared__ float sm[225];
  for (int i = threadIdx.x; i < 225; i += blockDim.x) sm[i] = res[b * 225 + i];
  __syncthreads();
  if (threadIdx.x < 15) {                      // column (H) softmax
    int w = threadIdx.x;
    float mx = -1e30f;
    for (int h = 0; h < 15; h++) mx = fmaxf(mx, sm[h * 15 + w]);
    float s = 0.f;
    for (int h = 0; h < 15; h++) { float e = __expf(sm[h * 15 + w] - mx); sm[h * 15 + w] = e; s += e; }
    float inv = 1.f / s;
    for (int h = 0; h < 15; h++) sm[h * 15 + w] *= inv;
  }
  __syncthreads();
  if (threadIdx.x < 15) {                      // row (W) softmax of the result
    int h = threadIdx.x;
    float mx = -1e30f;
    for (int w = 0; w < 15; w++) mx = fmaxf(mx, sm[h * 15 + w]);
    float e[15], s = 0.f;
    for (int w = 0; w < 15; w++) { e[w] = __expf(sm[h * 15 + w] - mx); s += e[w]; }
    float inv = 1.f / s;
    for (int w = 0; w < 15; w++) out[b * 225 + h * 15 + w] = e[w] * inv;
  }
}

// ---------------------------------------------------------------------------
extern "C" void kernel_launch(void* const* d_in, const int* in_sizes, int n_in,
                              void* d_out, int out_size, void* d_ws, size_t ws_size,
                              hipStream_t stream) {
  const float* img    = (const float*)d_in[0];
  const float* face   = (const float*)d_in[1];
  const float* w_img  = (const float*)d_in[2];
  const float* b_img  = (const float*)d_in[3];
  const float* g_img  = (const float*)d_in[4];
  const float* be_img = (const float*)d_in[5];
  const float* m_img  = (const float*)d_in[6];
  const float* v_img  = (const float*)d_in[7];
  const float* w_face = (const float*)d_in[8];
  const float* b_face = (const float*)d_in[9];
  const float* g_face = (const float*)d_in[10];
  const float* be_face= (const float*)d_in[11];
  const float* m_face = (const float*)d_in[12];
  const float* v_face = (const float*)d_in[13];

  const int NIMG  = 64 * 512 * 28 * 28;   // 25,690,112
  const int NFACE = 64 * 512 * 14 * 14;   //  6,422,528
  const int NW    = 512 * 512 * 9;        //  2,359,296

  char* ws = (char*)d_ws;
  size_t off = 0;
  auto alloc = [&](size_t bytes) -> void* {
    void* p = ws + off;
    off = (off + bytes + 255) & ~(size_t)255;
    return p;
  };

  unsigned short* img_bf  = (unsigned short*)alloc((size_t)NIMG  * 2);
  unsigned short* face_bf = (unsigned short*)alloc((size_t)NFACE * 2);
  unsigned short* wimg_bf = (unsigned short*)alloc((size_t)NW    * 2);
  unsigned short* wfac_bf = (unsigned short*)alloc((size_t)NW    * 2);
  float* x_buf   = (float*)alloc((size_t)NIMG  * 4);
  float* f_buf   = (float*)alloc((size_t)NFACE * 4);
  float* res_raw = (float*)alloc(64 * 225 * 4);
  float* bnAi = (float*)alloc(512 * 4);
  float* bnBi = (float*)alloc(512 * 4);
  float* bnAf = (float*)alloc(512 * 4);
  float* bnBf = (float*)alloc(512 * 4);

  // 1) fp32 -> bf16 conversions; weights permuted to tap-major [O][9][I]
  cvt_f32_bf16<<<(NIMG  + 255) / 256, 256, 0, stream>>>(img,  img_bf,  NIMG);
  cvt_f32_bf16<<<(NFACE + 255) / 256, 256, 0, stream>>>(face, face_bf, NFACE);
  cvt_weight_tapmajor<<<(NW + 255) / 256, 256, 0, stream>>>(w_img,  wimg_bf, NW);
  cvt_weight_tapmajor<<<(NW + 255) / 256, 256, 0, stream>>>(w_face, wfac_bf, NW);

  // 2) fold bias + eval BN into per-channel affine
  prep_bn<<<1, 512, 0, stream>>>(b_img,  g_img,  be_img,  m_img,  v_img,  bnAi, bnBi);
  prep_bn<<<1, 512, 0, stream>>>(b_face, g_face, be_face, m_face, v_face, bnAf, bnBf);

  // 3) WMMA implicit-GEMM convs (+ affine + ReLU)
  {
    dim3 grid(512 / BN, (64 * 28 * 28) / BM);   // (2, 392)
    conv3x3_bn_relu_wmma<28, 28><<<grid, 256, 0, stream>>>(img_bf, wimg_bf,
                                                           bnAi, bnBi, x_buf, 64);
  }
  {
    dim3 grid(512 / BN, (64 * 14 * 14) / BM);   // (2, 98)
    conv3x3_bn_relu_wmma<14, 14><<<grid, 256, 0, stream>>>(face_bf, wfac_bf,
                                                           bnAf, bnBf, f_buf, 64);
  }

  // 4) per-sample correlation
  {
    dim3 grid(15, 64);
    corr_kernel<<<grid, 256, 0, stream>>>(x_buf, f_buf, res_raw);
  }

  // 5) softmax over H then W
  softmax_hw<<<64, 32, 0, stream>>>(res_raw, (float*)d_out);
}
